// ReuseEncoderLayer_38706245272238
// MI455X (gfx1250) — matrix operand order repaired and checked by reading, not compiled, not run on hardware
//
#include <hip/hip_runtime.h>

// ViT encoder layer for MI455X (gfx1250, wave32, WMMA, TDM).
// B=64, N=197, D=768, H=12, hd=64, DFF=3072.
// GEMMs: v_wmma_f32_16x16x32_bf16, BK=64, double-buffered LDS staging.
// Tile staging uses the Tensor Data Mover (tensor_load_to_lds, 6-arg clang-23
// form) when available (wave0 issues DMA, s_wait_tensorcnt before barrier),
// else falls back to global_load_b128 -> ds_store_b128.
// Attention fused per (b,h) head entirely in LDS (~147KB/WG).

#define B_    64
#define NSEQ  197
#define D_    768
#define H_    12
#define HD_   64
#define DFF_  3072
#define MROWS (B_ * NSEQ)        // 12608 = 197 * 64

typedef __attribute__((ext_vector_type(16))) __bf16 bf16x16;
typedef __attribute__((ext_vector_type(8)))  float  f32x8;
typedef __attribute__((ext_vector_type(4)))  unsigned int v4u;
typedef __attribute__((ext_vector_type(8)))  int  v8i;
typedef __attribute__((ext_vector_type(4)))  int  v4i;

union FragB16 { uint4 u[2]; bf16x16 v; };

#if defined(__has_builtin)
#if __has_builtin(__builtin_amdgcn_tensor_load_to_lds) && \
    __has_builtin(__builtin_amdgcn_s_wait_tensorcnt)
#define USE_TDM 1
#endif
#endif
#ifndef USE_TDM
#define USE_TDM 0
#endif

__device__ __forceinline__ unsigned short f32_to_bf16(float f) {
  unsigned int u = __float_as_uint(f);
  unsigned int r = u + 0x7FFFu + ((u >> 16) & 1u);
  return (unsigned short)(r >> 16);
}

__device__ __forceinline__ f32x8 zero8() {
  f32x8 z;
  #pragma unroll
  for (int i = 0; i < 8; i++) z[i] = 0.0f;
  return z;
}

#if USE_TDM
// Issue one TDM 2D tile load: tileH rows x tileW bf16 elements, row stride
// strideElems (elements), packed contiguously into LDS at lds_off.
__device__ __forceinline__ void tdm_load_2d(unsigned lds_off, const void* gptr,
                                            int tileW, int tileH, int strideElems) {
  unsigned long long ga = (unsigned long long)(uintptr_t)gptr;
  v4u g0;
  g0[0] = 1u;                                            // count=1, user mode
  g0[1] = lds_off;                                       // lds_addr
  g0[2] = (unsigned)(ga & 0xFFFFFFFFu);                  // global_addr[31:0]
  g0[3] = (unsigned)((ga >> 32) & 0x1FFFFFFu) | (2u << 30); // addr[56:32] | type=2
  v8i g1;
  g1[0] = (int)(1u << 16);                               // data_size=1 (2B)
  g1[1] = (int)(((unsigned)tileW & 0xFFFFu) << 16);      // tensor_dim0[15:0]
  g1[2] = (int)((((unsigned)tileW >> 16) & 0xFFFFu) |
                (((unsigned)tileH & 0xFFFFu) << 16));    // dim0[31:16] | dim1[15:0]
  g1[3] = (int)((((unsigned)tileH >> 16) & 0xFFFFu) |
                (((unsigned)tileW & 0xFFFFu) << 16));    // dim1[31:16] | tile_dim0
  g1[4] = (int)((unsigned)tileH & 0xFFFFu);              // tile_dim1 (tile_dim2=0)
  g1[5] = strideElems;                                   // tensor_dim0_stride[31:0]
  g1[6] = 0; g1[7] = 0;
  v4i z4; z4[0] = 0; z4[1] = 0; z4[2] = 0; z4[3] = 0;
  v8i z8;
  #pragma unroll
  for (int i = 0; i < 8; i++) z8[i] = 0;
  // 6-arg form (clang-23 / therock): (g0, g1, g2, g3, g4, cpol)
  __builtin_amdgcn_tensor_load_to_lds(g0, g1, z4, z4, z8, 0);
}
#endif

// ---------------------------------------------------------------- utilities
// fp32 [K,N] -> bf16 [N,K] tiled transpose (one-time weight prep).
__global__ __launch_bounds__(256) void f2bfT_kernel(const float* __restrict__ in,
                                                    unsigned short* __restrict__ out,
                                                    int K, int N) {
  __shared__ float t[32][33];
  const int k0 = blockIdx.y * 32, n0 = blockIdx.x * 32;
  const int tx = threadIdx.x & 31, ty = threadIdx.x >> 5;
  for (int r = ty; r < 32; r += 8)
    t[r][tx] = in[(size_t)(k0 + r) * N + n0 + tx];
  __syncthreads();
  for (int r = ty; r < 32; r += 8)
    out[(size_t)(n0 + r) * K + k0 + tx] = f32_to_bf16(t[tx][r]);
}

// LayerNorm over last dim (D=768), one block per row, bf16 output.
__global__ __launch_bounds__(256) void ln_bf16_kernel(const float* __restrict__ x,
                                                      const float* __restrict__ g,
                                                      const float* __restrict__ b,
                                                      unsigned short* __restrict__ out) {
  const int row = blockIdx.x;
  const float* xr = x + (size_t)row * D_;
  __shared__ float red[256];
  const int tid = threadIdx.x;

  float s = 0.f;
  for (int i = tid; i < D_; i += 256) s += xr[i];
  red[tid] = s; __syncthreads();
  for (int off = 128; off > 0; off >>= 1) {
    if (tid < off) red[tid] += red[tid + off];
    __syncthreads();
  }
  const float mu = red[0] / (float)D_;
  __syncthreads();

  float v = 0.f;
  for (int i = tid; i < D_; i += 256) { float d = xr[i] - mu; v += d * d; }
  red[tid] = v; __syncthreads();
  for (int off = 128; off > 0; off >>= 1) {
    if (tid < off) red[tid] += red[tid + off];
    __syncthreads();
  }
  const float rstd = rsqrtf(red[0] / (float)D_ + 1e-5f);

  unsigned short* orow = out + (size_t)row * D_;
  for (int i = tid; i < D_; i += 256)
    orow[i] = f32_to_bf16((xr[i] - mu) * rstd * g[i] + b[i]);
}

// ---------------------------------------------------------------- GEMM
// C[M,N] = A[M,K](bf16) * Wt[N,K](bf16, pre-transposed) + bias.
//  EPI 0: outB = bf16((acc+bias)*scale)
//  EPI 1: outB = bf16(gelu_exact(acc+bias))
//  EPI 2: outF = resid + acc + bias
#define BM 64
#define BN 128
#define BK 64

template <int EPI>
__global__ __launch_bounds__(256)
void gemm_bf16_kernel(const unsigned short* __restrict__ A,
                      const unsigned short* __restrict__ Wt,
                      const float* __restrict__ bias,
                      const float* __restrict__ resid,
                      float scale,
                      unsigned short* __restrict__ outB,
                      float* __restrict__ outF,
                      int M, int N, int K) {
  __shared__ __align__(16) uint4 sA4[2 * (BM * BK / 8)];   // 2 x 8KB
  __shared__ __align__(16) uint4 sB4[2 * (BN * BK / 8)];   // 2 x 16KB

  const int tid  = threadIdx.x;
  const int wave = tid >> 5;
  const int lane = tid & 31;
  const int half = lane >> 4;
  const int l16  = lane & 15;
  const int m0 = blockIdx.y * BM;
  const int n0 = blockIdx.x * BN;
  const int mW = (wave >> 2) * 32;  // 0 / 32
  const int nW = (wave & 3) * 32;   // 0..96

  f32x8 acc[2][2];
  #pragma unroll
  for (int i = 0; i < 2; i++)
    #pragma unroll
    for (int j = 0; j < 2; j++) acc[i][j] = zero8();

  const unsigned int* sA32 = (const unsigned int*)sA4;
  const unsigned int* sB32 = (const unsigned int*)sB4;
  const uint4* Ag4 = (const uint4*)A;
  const uint4* Wg4 = (const uint4*)Wt;
  const int K8 = K >> 3;
  const int KT = K / BK;

  // ---- tile staging (TDM when available, else vector load/store)
  auto stage = [&](int buf, int k0) {
#if USE_TDM
    if (wave == 0) {
      tdm_load_2d((unsigned)(uintptr_t)&sA4[buf * (BM * BK / 8)],
                  A + (size_t)m0 * K + k0, BK, BM, K);
      tdm_load_2d((unsigned)(uintptr_t)&sB4[buf * (BN * BK / 8)],
                  Wt + (size_t)n0 * K + k0, BK, BN, K);
    }
#else
    #pragma unroll
    for (int e = tid; e < BM * BK / 8; e += 256) {       // 512 uint4
      int m = e >> 3, c = e & 7;
      sA4[buf * (BM * BK / 8) + e] = Ag4[(size_t)(m0 + m) * K8 + (k0 >> 3) + c];
    }
    #pragma unroll
    for (int e = tid; e < BN * BK / 8; e += 256) {       // 1024 uint4
      int n = e >> 3, c = e & 7;
      sB4[buf * (BN * BK / 8) + e] = Wg4[(size_t)(n0 + n) * K8 + (k0 >> 3) + c];
    }
#endif
  };

  stage(0, 0);
  for (int kt = 0; kt < KT; kt++) {
    const int buf = kt & 1;
#if USE_TDM
    if (wave == 0) __builtin_amdgcn_s_wait_tensorcnt((short)0);
#endif
    __syncthreads();
    if (kt + 1 < KT) stage(buf ^ 1, (kt + 1) * BK);

    const unsigned int* aBase = sA32 + buf * (BM * BK / 2);
    const unsigned int* bBase = sB32 + buf * (BN * BK / 2);
    #pragma unroll
    for (int kk = 0; kk < BK; kk += 32) {
      FragB16 fa[2], fb[2];
      #pragma unroll
      for (int mt = 0; mt < 2; mt++) {
        const unsigned int* p = aBase + (mW + mt * 16 + l16) * (BK / 2) + (kk >> 1);
        fa[mt].u[0] = *(const uint4*)(p + half * 4);
        fa[mt].u[1] = *(const uint4*)(p + 8 + half * 4);
      }
      #pragma unroll
      for (int nt = 0; nt < 2; nt++) {
        const unsigned int* p = bBase + (nW + nt * 16 + l16) * (BK / 2) + (kk >> 1);
        fb[nt].u[0] = *(const uint4*)(p + half * 4);
        fb[nt].u[1] = *(const uint4*)(p + 8 + half * 4);
      }
      #pragma unroll
      for (int mt = 0; mt < 2; mt++)
        #pragma unroll
        for (int nt = 0; nt < 2; nt++)
          acc[mt][nt] = __builtin_amdgcn_wmma_f32_16x16x32_bf16(
              false, fa[mt].v, false, fb[nt].v, (short)0, acc[mt][nt], false, false);
    }
    __syncthreads();
  }

  #pragma unroll
  for (int mt = 0; mt < 2; mt++)
    #pragma unroll
    for (int nt = 0; nt < 2; nt++) {
      const int col = n0 + nW + nt * 16 + l16;
      const float bv = bias[col];
      #pragma unroll
      for (int r = 0; r < 8; r++) {
        const int row = m0 + mW + mt * 16 + half * 8 + r;
        const size_t idx = (size_t)row * N + col;
        const float va = acc[mt][nt][r] + bv;
        if (EPI == 0) {
          outB[idx] = f32_to_bf16(va * scale);
        } else if (EPI == 1) {
          outB[idx] = f32_to_bf16(0.5f * va * (1.0f + erff(va * 0.70710678118654752f)));
        } else {
          outF[idx] = resid[idx] + va;
        }
      }
    }
}

// ---------------------------------------------------------------- attention
// One block per (b,h). K/V staged once; 64-query-row chunks:
// scores (WMMA) -> softmax (LDS fp32) -> probs bf16 -> P*V (WMMA).
#define NKP 208   // 13*16 key tiles
#define KP  224   // K-dim pad (multiple of 32) for P*V

__global__ __launch_bounds__(256)
void attention_kernel(const unsigned short* __restrict__ Q,   // [M,768] bf16, pre-scaled
                      const unsigned short* __restrict__ Kb,  // [M,768] bf16
                      const unsigned short* __restrict__ Vb,  // [M,768] bf16
                      float* __restrict__ preProj,            // [M,768] f32 (d_out)
                      unsigned short* __restrict__ attnB,     // [M,768] bf16 (ws)
                      float* __restrict__ attnW) {            // [B,H,197,197] f32 (d_out)
  __shared__ __align__(16) unsigned short sQ[64 * 64];
  __shared__ __align__(16) unsigned short sK[NKP * 64];   // Bt[j][d]
  __shared__ __align__(16) unsigned short sV[64 * KP];    // Bt[d][j]
  __shared__ __align__(16) float          sS[64 * NKP];
  __shared__ __align__(16) unsigned short sP[64 * KP];
  __shared__ float sRed[512];

  const int bh = blockIdx.x;
  const int b = bh / H_, h = bh % H_;
  const int tid  = threadIdx.x;
  const int wave = tid >> 5;
  const int lane = tid & 31;
  const int half = lane >> 4;
  const int l16  = lane & 15;

  unsigned int* sQu = (unsigned int*)sQ;
  unsigned int* sKu = (unsigned int*)sK;
  unsigned int* sVu = (unsigned int*)sV;
  unsigned int* sPu = (unsigned int*)sP;
  const unsigned int* Qu = (const unsigned int*)Q;
  const unsigned int* Ku = (const unsigned int*)Kb;

  // stage K as Bt[j][d] (row-contiguous copy), zero pad rows 197..207
  for (int e = tid; e < NKP * 32; e += 256) {       // dwords (2 bf16 each)
    int j = e >> 5, kp = e & 31;
    sKu[e] = (j < NSEQ) ? Ku[(size_t)(b * NSEQ + j) * (D_ / 2) + h * 32 + kp] : 0u;
  }
  // stage V transposed: sV[d][j] = V[b,j,h,d]; zero pad j>=197
  for (int e = tid; e < NKP * 64; e += 256) {
    int d = e & 63, j = e >> 6;
    unsigned short v = (j < NSEQ) ? Vb[(size_t)(b * NSEQ + j) * D_ + h * HD_ + d] : (unsigned short)0;
    sV[d * KP + j] = v;
  }
  for (int e = tid; e < 64 * (KP - NKP); e += 256) {
    int d = e / (KP - NKP), j = NKP + e % (KP - NKP);
    sV[d * KP + j] = 0;
  }

  for (int c = 0; c < 4; c++) {
    const int r0 = c * 64;
    const int mTiles = (13 - c * 4) < 4 ? (13 - c * 4) : 4;

    // stage Q chunk
    for (int e = tid; e < 64 * 32; e += 256) {      // dwords
      int i = r0 + (e >> 5), kp = e & 31;
      sQu[e] = (i < NSEQ) ? Qu[(size_t)(b * NSEQ + i) * (D_ / 2) + h * 32 + kp] : 0u;
    }
    __syncthreads();

    // ---- scores: S = Q * K^T (K=64 -> 2 WMMA steps)
    for (int t = wave; t < mTiles * 13; t += 8) {
      const int mt = t / 13, nt = t % 13;
      f32x8 acc = zero8();
      #pragma unroll
      for (int kk = 0; kk < 64; kk += 32) {
        FragB16 fa, fb;
        const unsigned int* pa = sQu + (mt * 16 + l16) * 32 + (kk >> 1);
        fa.u[0] = *(const uint4*)(pa + half * 4);
        fa.u[1] = *(const uint4*)(pa + 8 + half * 4);
        const unsigned int* pb = sKu + (nt * 16 + l16) * 32 + (kk >> 1);
        fb.u[0] = *(const uint4*)(pb + half * 4);
        fb.u[1] = *(const uint4*)(pb + 8 + half * 4);
        acc = __builtin_amdgcn_wmma_f32_16x16x32_bf16(
            false, fa.v, false, fb.v, (short)0, acc, false, false);
      }
      #pragma unroll
      for (int r = 0; r < 8; r++)
        sS[(mt * 16 + half * 8 + r) * NKP + nt * 16 + l16] = acc[r];
    }
    __syncthreads();

    // ---- softmax over j=0..196 (4 threads per row)
    const int rows = mTiles * 16;
    const int row = tid >> 2, part = tid & 3;
    if (row < rows) {
      float mx = -1e30f;
      for (int j = part; j < NSEQ; j += 4) mx = fmaxf(mx, sS[row * NKP + j]);
      sRed[row * 4 + part] = mx;
    }
    __syncthreads();
    if (row < rows) {
      const float mx = fmaxf(fmaxf(sRed[row * 4], sRed[row * 4 + 1]),
                             fmaxf(sRed[row * 4 + 2], sRed[row * 4 + 3]));
      float s = 0.f;
      for (int j = part; j < NSEQ; j += 4) {
        float e = __expf(sS[row * NKP + j] - mx);
        sS[row * NKP + j] = e;
        s += e;
      }
      sRed[256 + row * 4 + part] = s;
    }
    __syncthreads();
    if (row < rows) {
      const float tot = sRed[256 + row * 4] + sRed[256 + row * 4 + 1] +
                        sRed[256 + row * 4 + 2] + sRed[256 + row * 4 + 3];
      const float inv = 1.0f / tot;
      const bool valid = (r0 + row) < NSEQ;
      for (int j = part; j < KP; j += 4) {
        unsigned short pb = 0;
        if (j < NSEQ && valid) {
          const float p = sS[row * NKP + j] * inv;
          attnW[((size_t)bh * NSEQ + (r0 + row)) * NSEQ + j] = p;
          pb = f32_to_bf16(p);
        }
        sP[row * KP + j] = pb;
      }
    }
    __syncthreads();

    // ---- out = P * V (K=224 -> 7 WMMA steps)
    for (int t = wave; t < mTiles * 4; t += 8) {
      const int mt = t >> 2, nt = t & 3;
      f32x8 acc = zero8();
      #pragma unroll
      for (int kk = 0; kk < KP; kk += 32) {
        FragB16 fa, fb;
        const unsigned int* pa = sPu + (mt * 16 + l16) * (KP / 2) + (kk >> 1);
        fa.u[0] = *(const uint4*)(pa + half * 4);
        fa.u[1] = *(const uint4*)(pa + 8 + half * 4);
        const unsigned int* pb = sVu + (nt * 16 + l16) * (KP / 2) + (kk >> 1);
        fb.u[0] = *(const uint4*)(pb + half * 4);
        fb.u[1] = *(const uint4*)(pb + 8 + half * 4);
        acc = __builtin_amdgcn_wmma_f32_16x16x32_bf16(
            false, fa.v, false, fb.v, (short)0, acc, false, false);
      }
      #pragma unroll
      for (int r = 0; r < 8; r++) {
        const int i = r0 + mt * 16 + half * 8 + r;
        if (i < NSEQ) {
          const size_t idx = (size_t)(b * NSEQ + i) * D_ + h * HD_ + nt * 16 + l16;
          preProj[idx] = acc[r];
          attnB[idx] = f32_to_bf16(acc[r]);
        }
      }
    }
    __syncthreads();
  }
}

// ---------------------------------------------------------------- launch
extern "C" void kernel_launch(void* const* d_in, const int* in_sizes, int n_in,
                              void* d_out, int out_size, void* d_ws, size_t ws_size,
                              hipStream_t stream) {
  (void)in_sizes; (void)n_in; (void)out_size; (void)ws_size;
  const float* x     = (const float*)d_in[0];
  const float* ln1_g = (const float*)d_in[1];
  const float* ln1_b = (const float*)d_in[2];
  const float* wq = (const float*)d_in[3];  const float* bq = (const float*)d_in[4];
  const float* wk = (const float*)d_in[5];  const float* bk = (const float*)d_in[6];
  const float* wv = (const float*)d_in[7];  const float* bv = (const float*)d_in[8];
  const float* wo = (const float*)d_in[9];  const float* bo = (const float*)d_in[10];
  const float* ln2_g = (const float*)d_in[11];
  const float* ln2_b = (const float*)d_in[12];
  const float* w1 = (const float*)d_in[13]; const float* b1 = (const float*)d_in[14];
  const float* w2 = (const float*)d_in[15]; const float* b2 = (const float*)d_in[16];

  float* out      = (float*)d_out;
  float* preProj  = out;                                 // [M, D]
  float* hidden   = out + (size_t)MROWS * D_;            // [M, D]
  float* attnW    = out + (size_t)2 * MROWS * D_;        // [B,H,N,N]

  char* ws = (char*)d_ws;
  size_t off = 0;
  auto alloc = [&](size_t bytes) -> void* {
    void* p = ws + off;
    off = (off + bytes + 255) & ~(size_t)255;
    return p;
  };
  const size_t DD = (size_t)D_ * D_, DF = (size_t)D_ * DFF_;
  // all weight buffers hold W^T as bf16 [N,K]
  unsigned short* wqB = (unsigned short*)alloc(DD * 2);
  unsigned short* wkB = (unsigned short*)alloc(DD * 2);
  unsigned short* wvB = (unsigned short*)alloc(DD * 2);
  unsigned short* woB = (unsigned short*)alloc(DD * 2);
  unsigned short* w1B = (unsigned short*)alloc(DF * 2);
  unsigned short* w2B = (unsigned short*)alloc(DF * 2);
  unsigned short* attnB = (unsigned short*)alloc((size_t)MROWS * D_ * 2);
  float*          hsF   = (float*)alloc((size_t)MROWS * D_ * 4);
  unsigned short* ln2B  = (unsigned short*)alloc((size_t)MROWS * D_ * 2);
  // region R: hB,qB,kB,vB -- dead after attention; gelu buffer aliases it
  unsigned short* hB = (unsigned short*)(ws + off);
  unsigned short* qB = hB + (size_t)MROWS * D_;
  unsigned short* kB = qB + (size_t)MROWS * D_;
  unsigned short* vB = kB + (size_t)MROWS * D_;
  unsigned short* gB = hB;   // [M, DFF] bf16, aliases h/q/k/v (exact size match)

  // 1) weights -> transposed bf16 [N,K]
  dim3 gT1(D_ / 32, D_ / 32);
  f2bfT_kernel<<<gT1, 256, 0, stream>>>(wq, wqB, D_, D_);
  f2bfT_kernel<<<gT1, 256, 0, stream>>>(wk, wkB, D_, D_);
  f2bfT_kernel<<<gT1, 256, 0, stream>>>(wv, wvB, D_, D_);
  f2bfT_kernel<<<gT1, 256, 0, stream>>>(wo, woB, D_, D_);
  dim3 gT2(DFF_ / 32, D_ / 32);
  f2bfT_kernel<<<gT2, 256, 0, stream>>>(w1, w1B, D_, DFF_);
  dim3 gT3(D_ / 32, DFF_ / 32);
  f2bfT_kernel<<<gT3, 256, 0, stream>>>(w2, w2B, DFF_, D_);

  // 2) LN1
  ln_bf16_kernel<<<MROWS, 256, 0, stream>>>(x, ln1_g, ln1_b, hB);

  // 3) QKV projections (scale 1/8 folded into Q)
  dim3 gQKV(D_ / BN, MROWS / BM);
  gemm_bf16_kernel<0><<<gQKV, 256, 0, stream>>>(hB, wqB, bq, nullptr, 0.125f, qB, nullptr, MROWS, D_, D_);
  gemm_bf16_kernel<0><<<gQKV, 256, 0, stream>>>(hB, wkB, bk, nullptr, 1.0f,   kB, nullptr, MROWS, D_, D_);
  gemm_bf16_kernel<0><<<gQKV, 256, 0, stream>>>(hB, wvB, bv, nullptr, 1.0f,   vB, nullptr, MROWS, D_, D_);

  // 4) fused attention (scores+softmax+P*V per head in LDS)
  attention_kernel<<<B_ * H_, 256, 0, stream>>>(qB, kB, vB, preProj, attnB, attnW);

  // 5) hs = x + pre_proj @ wo + bo
  gemm_bf16_kernel<2><<<gQKV, 256, 0, stream>>>(attnB, woB, bo, x, 1.0f, nullptr, hsF, MROWS, D_, D_);

  // 6) LN2
  ln_bf16_kernel<<<MROWS, 256, 0, stream>>>(hsF, ln2_g, ln2_b, ln2B);

  // 7) MLP1: gelu(ln2 @ w1 + b1)  -> gB
  dim3 gM1(DFF_ / BN, MROWS / BM);
  gemm_bf16_kernel<1><<<gM1, 256, 0, stream>>>(ln2B, w1B, b1, nullptr, 1.0f, gB, nullptr, MROWS, DFF_, D_);

  // 8) hidden = hs + gB @ w2 + b2
  gemm_bf16_kernel<2><<<gQKV, 256, 0, stream>>>(gB, w2B, b2, hsF, 1.0f, nullptr, hidden, MROWS, D_, DFF_);
}
